// Qwen3VLVisionAttention_49675591746245
// MI455X (gfx1250) — matrix operand order repaired and checked by reading, not compiled, not run on hardware
//
#include <hip/hip_runtime.h>
#include <hip/hip_bf16.h>
#include <stdint.h>

// Problem constants (match reference)
#define SEQ 2304
#define HID 1024
#define NH  16
#define HD  64
#define HID3 (3 * HID)

// ---------------------------------------------------------------------------
// WMMA vector types (gfx1250, wave32)
// ---------------------------------------------------------------------------
typedef __bf16 v16bf __attribute__((ext_vector_type(16)));
typedef float  v8f   __attribute__((ext_vector_type(8)));
typedef int    v8i   __attribute__((ext_vector_type(8)));
typedef int    v4i   __attribute__((ext_vector_type(4)));

#define AS1 __attribute__((address_space(1)))
#define AS3 __attribute__((address_space(3)))

// Probe the gfx1250 async global->LDS path; fall back to sync 16B copies.
#if defined(__has_builtin)
#  if __has_builtin(__builtin_amdgcn_global_load_async_to_lds_b128) && \
      __has_builtin(__builtin_amdgcn_s_wait_asynccnt)
#    define USE_ASYNC 1
#  endif
#endif
#ifndef USE_ASYNC
#  define USE_ASYNC 0
#endif

// 16-byte global -> LDS copy (async on CDNA5 when available).
// Builtin signature (from hipcc diagnostic): param0 = v4i AS1* (global src),
// param1 = v4i AS3* (LDS dst), then imm offset, imm cpol.
__device__ __forceinline__ void copy16_g2lds(void* lds, const void* g) {
#if USE_ASYNC
    __builtin_amdgcn_global_load_async_to_lds_b128(
        (AS1 v4i*)(uintptr_t)g,
        (AS3 v4i*)(unsigned)(uintptr_t)lds,
        /*offset=*/0, /*cpol=*/0);
#else
    *(uint4*)lds = *(const uint4*)g;
#endif
}
__device__ __forceinline__ void g2lds_drain() {
#if USE_ASYNC
    __builtin_amdgcn_s_wait_asynccnt(0);
#endif
}

// f32 -> bf16 (round to nearest even), stored as raw u16
__device__ __forceinline__ unsigned short f32_to_bf16(float f) {
    unsigned int u = __float_as_uint(f);
    u = (u + 0x7FFFu + ((u >> 16) & 1u)) >> 16;
    return (unsigned short)u;
}

// A-matrix (16x32 bf16) per-lane K offset for VGPR v:
//   lanes 0-15: K = (v>>2)*16 + (v&3)*2 ; lanes 16-31: +8 within each 16-group
__device__ __forceinline__ int a_koff(int lane, int v) {
    return ((v >> 2) << 4) + (((lane >> 4) & 1) << 3) + ((v & 3) << 1);
}
// B-matrix (32x16 bf16) per-lane K offset for VGPR v (same order as A):
//   lanes 0-15: K = (v>>2)*16 + (v&3)*2 ; lanes 16-31: +8 within each 16-group
__device__ __forceinline__ int b_koff(int lane, int v) {
    return ((v >> 2) << 4) + (((lane >> 4) & 1) << 3) + ((v & 3) << 1);
}

template <int STRIDE>
__device__ __forceinline__ v16bf load_frag_a(const unsigned short* base, int row,
                                             int kk, int lane) {
    const unsigned short* p = base + row * STRIDE + kk;
    v8i r;
#pragma unroll
    for (int v = 0; v < 8; ++v)
        r[v] = *(const unsigned int*)(p + a_koff(lane, v));
    return __builtin_bit_cast(v16bf, r);
}

template <int STRIDE>
__device__ __forceinline__ v16bf load_frag_b(const unsigned short* base, int row,
                                             int kk, int lane) {
    const unsigned short* p = base + row * STRIDE + kk;
    v8i r;
#pragma unroll
    for (int v = 0; v < 8; ++v)
        r[v] = *(const unsigned int*)(p + b_koff(lane, v));
    return __builtin_bit_cast(v16bf, r);
}

// ---------------------------------------------------------------------------
// Kernel: f32 -> bf16 convert
// ---------------------------------------------------------------------------
__global__ __launch_bounds__(256) void cvt_f32_bf16_kernel(
    const float* __restrict__ in, unsigned short* __restrict__ out, int n) {
    int i = blockIdx.x * blockDim.x + threadIdx.x;
    if (i < n) out[i] = f32_to_bf16(in[i]);
}

// ---------------------------------------------------------------------------
// Kernel: bf16 WMMA GEMM  C[M,N](f32) = A[M,K](bf16) @ B[K,N](bf16) + bias[N]
// 64x64 tile, BK=32, 128 threads = 4 waves; wave w owns rows 16w..16w+15.
// ---------------------------------------------------------------------------
__global__ __launch_bounds__(128) void gemm_bf16_kernel(
    const unsigned short* __restrict__ A, const unsigned short* __restrict__ B,
    const float* __restrict__ bias, float* __restrict__ C, int M, int N, int K) {
    __shared__ __align__(16) unsigned short lA[64][32];   // [m][k]
    __shared__ __align__(16) unsigned short lBt[64][32];  // [n][k] (B transposed)

    const int tid = threadIdx.x;
    const int wave = tid >> 5, lane = tid & 31;
    const int m0 = blockIdx.y * 64, n0 = blockIdx.x * 64;

    v8f acc[4] = {};

    for (int k0 = 0; k0 < K; k0 += 32) {
        // A tile: 64 rows x 32 bf16 = 256 x 16B, async copy to LDS
        for (int u = tid; u < 256; u += 128) {
            int row = u >> 2, c8 = (u & 3) << 3;
            copy16_g2lds(&lA[row][c8], &A[(size_t)(m0 + row) * K + k0 + c8]);
        }
        // B tile transposed into [n][k] (scatter -> scalar DS stores)
        for (int e = tid; e < 2048; e += 128) {
            int kr = e >> 6, n = e & 63;
            lBt[n][kr] = B[(size_t)(k0 + kr) * N + n0 + n];
        }
        // prefetch next K-step tiles while this one is consumed
        if (k0 + 32 < K) {
            __builtin_prefetch(&A[(size_t)(m0 + (tid & 63)) * K + k0 + 32], 0, 3);
            __builtin_prefetch(&B[(size_t)(k0 + 32 + (tid & 31)) * N + n0], 0, 3);
        }
        g2lds_drain();
        __syncthreads();

        v16bf a = load_frag_a<32>(&lA[0][0], (wave << 4) + (lane & 15), 0, lane);
#pragma unroll
        for (int nt = 0; nt < 4; ++nt) {
            v16bf b = load_frag_b<32>(&lBt[0][0], (nt << 4) + (lane & 15), 0, lane);
            acc[nt] = __builtin_amdgcn_wmma_f32_16x16x32_bf16(
                false, a, false, b, (short)0, acc[nt], false, false);
        }
        __syncthreads();
    }

    // C layout: VGPR r -> (m = r + 8*(lane>>4), n = lane&15)
    const int mbase = m0 + (wave << 4) + ((lane >> 4) << 3);
#pragma unroll
    for (int nt = 0; nt < 4; ++nt) {
        int n = n0 + (nt << 4) + (lane & 15);
        float bv = bias[n];
#pragma unroll
        for (int r = 0; r < 8; ++r)
            C[(size_t)(mbase + r) * N + n] = acc[nt][r] + bv;
    }
}

// ---------------------------------------------------------------------------
// Kernel: RoPE on q,k + split qkv -> head-major bf16 Q/K/V  [NH][SEQ][HD]
// ---------------------------------------------------------------------------
__global__ __launch_bounds__(256) void rope_split_kernel(
    const float* __restrict__ qkv,   // [SEQ][3*HID]
    const float* __restrict__ cosb,  // [SEQ][HD]
    const float* __restrict__ sinb,  // [SEQ][HD]
    unsigned short* __restrict__ Qh, unsigned short* __restrict__ Kh,
    unsigned short* __restrict__ Vh) {
    int idx = blockIdx.x * blockDim.x + threadIdx.x;
    if (idx >= SEQ * NH * HD) return;
    int d = idx & (HD - 1);
    int h = (idx / HD) & (NH - 1);
    int s = idx / (NH * HD);

    size_t base = (size_t)s * HID3;
    float q = qkv[base + h * HD + d];
    float k = qkv[base + HID + h * HD + d];
    float v = qkv[base + 2 * HID + h * HD + d];
    float c = cosb[s * HD + d];
    float sn = sinb[s * HD + d];

    const int half = HD / 2;
    int dp = (d < half) ? d + half : d - half;
    float qp = qkv[base + h * HD + dp];
    float kp = qkv[base + HID + h * HD + dp];
    float qo = (d < half) ? (q * c - qp * sn) : (q * c + qp * sn);
    float ko = (d < half) ? (k * c - kp * sn) : (k * c + kp * sn);

    size_t o = ((size_t)h * SEQ + s) * HD + d;
    Qh[o] = f32_to_bf16(qo);
    Kh[o] = f32_to_bf16(ko);
    Vh[o] = f32_to_bf16(v);
}

// ---------------------------------------------------------------------------
// Kernel: flash attention, one (head, 64-query block) per workgroup.
// 128 threads = 4 waves; wave w owns queries 16w..16w+15 of the block.
// Online softmax; P re-laid out C->A through per-wave LDS; PV via WMMA.
// ---------------------------------------------------------------------------
__global__ __launch_bounds__(128) void flash_attn_kernel(
    const unsigned short* __restrict__ Qh,  // [NH][SEQ][HD]
    const unsigned short* __restrict__ Kh, const unsigned short* __restrict__ Vh,
    unsigned short* __restrict__ O) {  // [SEQ][HID] bf16
    __shared__ __align__(16) unsigned short lQ[64][64];     // [q][d]
    __shared__ __align__(16) unsigned short lK[64][64];     // [key][d]
    __shared__ __align__(16) unsigned short lVt[64][64];    // [d][key]
    __shared__ __align__(16) unsigned short lP[4][16][64];  // per-wave [m][key]

    const int tid = threadIdx.x;
    const int wave = tid >> 5, lane = tid & 31;
    const int h = blockIdx.y;
    const int q0 = blockIdx.x * 64;
    const size_t hb = (size_t)h * SEQ * HD;
    const float scale = 0.125f;  // HD^-0.5

    // Q tile: 64 rows x 64 bf16 = 512 x 16B, async copy
    for (int u = tid; u < 512; u += 128) {
        int row = u >> 3, c8 = (u & 7) << 3;
        copy16_g2lds(&lQ[row][c8], &Qh[hb + (size_t)(q0 + row) * HD + c8]);
    }

    v8f acc[4] = {};
    float mrow[8], lrow[8];
#pragma unroll
    for (int r = 0; r < 8; ++r) { mrow[r] = -1e30f; lrow[r] = 0.f; }

    for (int kb = 0; kb < SEQ; kb += 64) {
        for (int u = tid; u < 512; u += 128) {
            int row = u >> 3, c8 = (u & 7) << 3;
            copy16_g2lds(&lK[row][c8], &Kh[hb + (size_t)(kb + row) * HD + c8]);
        }
        for (int e = tid; e < 4096; e += 128) {
            int key = e >> 6, d = e & 63;
            lVt[d][key] = Vh[hb + (size_t)(kb + key) * HD + d];
        }
        if (kb + 64 < SEQ) {
            __builtin_prefetch(&Kh[hb + (size_t)(kb + 64 + (tid & 63)) * HD], 0, 3);
            __builtin_prefetch(&Vh[hb + (size_t)(kb + 64 + (tid & 63)) * HD], 0, 3);
        }
        g2lds_drain();
        __syncthreads();

        // S = (Q K^T) * scale : 16x64 per wave in 4 C-fragments
        v8f s[4] = {};
#pragma unroll
        for (int kk = 0; kk < 64; kk += 32) {
            v16bf a = load_frag_a<64>(&lQ[0][0], (wave << 4) + (lane & 15), kk, lane);
#pragma unroll
            for (int nt = 0; nt < 4; ++nt) {
                v16bf b = load_frag_b<64>(&lK[0][0], (nt << 4) + (lane & 15), kk, lane);
                s[nt] = __builtin_amdgcn_wmma_f32_16x16x32_bf16(
                    false, a, false, b, (short)0, s[nt], false, false);
            }
        }
#pragma unroll
        for (int nt = 0; nt < 4; ++nt) s[nt] *= scale;

        // Online softmax: rows live across the 16 lanes of each wave half
        float alpha[8];
#pragma unroll
        for (int r = 0; r < 8; ++r) {
            float mx = fmaxf(fmaxf(s[0][r], s[1][r]), fmaxf(s[2][r], s[3][r]));
#pragma unroll
            for (int msk = 1; msk < 16; msk <<= 1)
                mx = fmaxf(mx, __shfl_xor(mx, msk, 32));
            float mnew = fmaxf(mrow[r], mx);
            alpha[r] = __expf(mrow[r] - mnew);
            mrow[r] = mnew;
            float rsum = 0.f;
#pragma unroll
            for (int nt = 0; nt < 4; ++nt) {
                float p = __expf(s[nt][r] - mnew);
                s[nt][r] = p;
                rsum += p;
            }
#pragma unroll
            for (int msk = 1; msk < 16; msk <<= 1)
                rsum += __shfl_xor(rsum, msk, 32);
            lrow[r] = lrow[r] * alpha[r] + rsum;
        }

        // Stage P (bf16) into per-wave LDS; rescale running accumulator.
        // DS ops from one wave execute in order -> no barrier needed here.
        const int mh = (lane >> 4) << 3;
#pragma unroll
        for (int nt = 0; nt < 4; ++nt) {
#pragma unroll
            for (int r = 0; r < 8; ++r) {
                lP[wave][mh + r][(nt << 4) + (lane & 15)] = f32_to_bf16(s[nt][r]);
                acc[nt][r] *= alpha[r];
            }
        }

        // acc += P @ V  (A = P 16x64, B = V via transposed tile)
#pragma unroll
        for (int kk = 0; kk < 64; kk += 32) {
            v16bf a = load_frag_a<64>(&lP[wave][0][0], lane & 15, kk, lane);
#pragma unroll
            for (int nt = 0; nt < 4; ++nt) {
                v16bf b = load_frag_b<64>(&lVt[0][0], (nt << 4) + (lane & 15), kk, lane);
                acc[nt] = __builtin_amdgcn_wmma_f32_16x16x32_bf16(
                    false, a, false, b, (short)0, acc[nt], false, false);
            }
        }
        __syncthreads();
    }

    // Normalize and write attention output (bf16, [SEQ][HID] token-major)
    const int srow = q0 + (wave << 4) + ((lane >> 4) << 3);
#pragma unroll
    for (int nt = 0; nt < 4; ++nt) {
        int col = h * HD + (nt << 4) + (lane & 15);
#pragma unroll
        for (int r = 0; r < 8; ++r) {
            float v = acc[nt][r] / lrow[r];
            O[(size_t)(srow + r) * HID + col] = f32_to_bf16(v);
        }
    }
}

// ---------------------------------------------------------------------------
// Launch
// ---------------------------------------------------------------------------
extern "C" void kernel_launch(void* const* d_in, const int* in_sizes, int n_in,
                              void* d_out, int out_size, void* d_ws, size_t ws_size,
                              hipStream_t stream) {
    (void)in_sizes; (void)n_in; (void)out_size; (void)ws_size;
    const float* hs     = (const float*)d_in[0];
    const float* cosb   = (const float*)d_in[1];
    const float* sinb   = (const float*)d_in[2];
    const float* w_qkv  = (const float*)d_in[3];
    const float* b_qkv  = (const float*)d_in[4];
    const float* w_proj = (const float*)d_in[5];
    const float* b_proj = (const float*)d_in[6];
    float* out = (float*)d_out;

    char* ws = (char*)d_ws;
    size_t off = 0;
    auto alloc = [&](size_t bytes) -> void* {
        void* p = (void*)(ws + off);
        off += (bytes + 255) & ~(size_t)255;
        return p;
    };
    unsigned short* hs_bf    = (unsigned short*)alloc((size_t)SEQ * HID * 2);
    unsigned short* wqkv_bf  = (unsigned short*)alloc((size_t)HID * HID3 * 2);
    unsigned short* wproj_bf = (unsigned short*)alloc((size_t)HID * HID * 2);
    float*          qkv_f32  = (float*)alloc((size_t)SEQ * HID3 * 4);
    unsigned short* Qh       = (unsigned short*)alloc((size_t)NH * SEQ * HD * 2);
    unsigned short* Kh       = (unsigned short*)alloc((size_t)NH * SEQ * HD * 2);
    unsigned short* Vh       = (unsigned short*)alloc((size_t)NH * SEQ * HD * 2);
    unsigned short* attn_bf  = (unsigned short*)alloc((size_t)SEQ * HID * 2);

    // 1) f32 -> bf16 conversions
    cvt_f32_bf16_kernel<<<(SEQ * HID + 255) / 256, 256, 0, stream>>>(hs, hs_bf, SEQ * HID);
    cvt_f32_bf16_kernel<<<(HID * HID3 + 255) / 256, 256, 0, stream>>>(w_qkv, wqkv_bf, HID * HID3);
    cvt_f32_bf16_kernel<<<(HID * HID + 255) / 256, 256, 0, stream>>>(w_proj, wproj_bf, HID * HID);

    // 2) QKV projection: [2304,1024] @ [1024,3072] + b -> f32
    gemm_bf16_kernel<<<dim3(HID3 / 64, SEQ / 64), 128, 0, stream>>>(
        hs_bf, wqkv_bf, b_qkv, qkv_f32, SEQ, HID3, HID);

    // 3) RoPE + split to head-major bf16
    rope_split_kernel<<<(SEQ * NH * HD + 255) / 256, 256, 0, stream>>>(
        qkv_f32, cosb, sinb, Qh, Kh, Vh);

    // 4) Flash attention, one block per (64 queries, head)
    flash_attn_kernel<<<dim3(SEQ / 64, NH), 128, 0, stream>>>(Qh, Kh, Vh, attn_bf);

    // 5) Output projection: [2304,1024] @ [1024,1024] + b -> f32 out
    gemm_bf16_kernel<<<dim3(HID / 64, SEQ / 64), 128, 0, stream>>>(
        attn_bf, wproj_bf, b_proj, out, SEQ, HID, HID);
}
